// Model_88407606821077
// MI455X (gfx1250) — compile-verified, matrix operand order checked by reading, not run on hardware
//
#include <hip/hip_runtime.h>
#include <hip/hip_bf16.h>

// ---------------------------------------------------------------------------
// Transformer encoder (B=64,S=512,DIM=128,FFDIM=512,H=4,blocks=4) for gfx1250.
// bf16 WMMA (f32 accumulate) for all GEMM-shaped math; Tensor Data Mover
// stages weight panels into LDS; activations kept as bf16 shadows so WMMA
// fragments are raw b128 loads. Softmax/LayerNorm/head in f32 VALU.
// GEMM: 4 waves/WG, each wave 2 M-tiles x 4 N-tiles => 8 WMMA per K-step
// with every LDS B-fragment reused twice.
// ---------------------------------------------------------------------------

#define BB      64
#define SS      512
#define DIM     128
#define FFDIM   512
#define NHEADS  4
#define HD      32
#define NBLOCKS 4
#define BSROWS  (BB * SS)   // 32768
#define LN_EPS  1e-5f
#define NEG_INF (-1e30f)

typedef __attribute__((ext_vector_type(16))) __bf16    v16bf;
typedef __attribute__((ext_vector_type(8)))  __bf16    v8bf;
typedef __attribute__((ext_vector_type(8)))  float     v8f;
typedef __attribute__((ext_vector_type(4)))  unsigned  su32x4;
typedef __attribute__((ext_vector_type(8)))  unsigned  su32x8;

// 16-bit WMMA A/B fragment K-mapping (wave32):
//   lanes 0-15 : halves 0-7 -> K kb+0..7, halves 8-15 -> K kb+16..23 (kb=0)
//   lanes 16-31: same with kb=8.
// => per lane the fragment is two contiguous 8-half (16 B) groups.
__device__ __forceinline__ int frag_k(int lane, int h) {
    return ((h >> 3) << 4) | ((lane & 16) >> 1) | (h & 7);
}

// Load one 16-bit fragment: two contiguous 16-byte groups at p and p+16 halves.
__device__ __forceinline__ v16bf load_frag(const __bf16* p) {
    v8bf lo = *(const v8bf*)p;
    v8bf hi = *(const v8bf*)(p + 16);
    return __builtin_shufflevector(lo, hi, 0, 1, 2, 3, 4, 5, 6, 7,
                                           8, 9, 10, 11, 12, 13, 14, 15);
}

// ---------------------------------------------------------------------------
// f32 -> bf16 bulk convert (weights, one-shot per launch).
// ---------------------------------------------------------------------------
__global__ __launch_bounds__(256)
void cvt_bf16_kernel(const float* __restrict__ src, __bf16* __restrict__ dst, int n) {
    const int i = blockIdx.x * 256 + threadIdx.x;
    if (i < n) dst[i] = (__bf16)src[i];
}

// ---------------------------------------------------------------------------
// Embedding + overwrite channels 0/1; writes f32 feat and bf16 shadow.
// ---------------------------------------------------------------------------
__global__ __launch_bounds__(DIM)
void embed_kernel(const int* __restrict__ tok,
                  const int* __restrict__ picked,
                  const int* __restrict__ skipped,
                  const int* __restrict__ nup,
                  const float* __restrict__ emb,
                  float* __restrict__ feat,
                  __bf16* __restrict__ featbf) {
    const int idx = blockIdx.x;          // b*S + s
    const int d   = threadIdx.x;
    const int b   = idx / SS;
    const int s   = idx - b * SS;
    float v = emb[(size_t)tok[idx] * DIM + d];
    if (d == 0) v = (s >= picked[b] + skipped[b]) ? 1.f : 0.f;
    if (d == 1) v = (nup[idx] > 0) ? 1.f : 0.f;
    feat[(size_t)idx * DIM + d]   = v;
    featbf[(size_t)idx * DIM + d] = (__bf16)v;
}

// ---------------------------------------------------------------------------
// Y = act(A @ W^T + bias).  A: bf16 [M x lda], W: bf16 (N x K row-major).
// Workgroup = 4 waves, 128-row M-panel x 64-col N-panel. The N-panel of W is
// DMA'd into LDS by the Tensor Data Mover in 64x128 chunks. Each wave owns
// two M-tiles so every LDS B-fragment feeds two WMMAs. OUT = float or __bf16.
// ---------------------------------------------------------------------------
template <bool RELU, typename OUT>
__global__ __launch_bounds__(128)
void gemm_bf16(const __bf16* __restrict__ A, int lda,
               const __bf16* __restrict__ W,
               const float* __restrict__ bias,
               OUT* __restrict__ Y, int ldy,
               int K) {
    __shared__ __bf16 wpan[64 * 128];    // 16 KB: 64 N-rows x 128 K bf16

    const int lane  = threadIdx.x & 31;
    const int wave  = threadIdx.x >> 5;
    const int m0    = (blockIdx.x * 4 + wave) * 32;   // 2 M-tiles per wave
    const int n0    = blockIdx.y * 64;
    const int mrow0 = m0 + (lane & 15);
    const int mrow1 = mrow0 + 16;
    const int ncol  = lane & 15;
    const int kbase = (lane & 16) >> 1;  // 0 or 8
    const int rbase = kbase;             // C row base for this half-wave

    v8f acc0[4] = {};
    v8f acc1[4] = {};
    for (int kc = 0; kc < K; kc += 128) {
        __syncthreads();                 // protect previous chunk's readers
        if (wave == 0) {
            // ---- Tensor DMA descriptor (D#): 2D tile, 64 rows x 128 bf16 ----
            const unsigned long long ga =
                (unsigned long long)(uintptr_t)(W + (size_t)n0 * K + kc);
            su32x4 g0;
            g0.x = 1u;                                 // count=1, user mode
            g0.y = 0u;                                 // lds_addr (wpan @ 0)
            g0.z = (unsigned)ga;                       // global_addr[31:0]
            g0.w = (unsigned)(ga >> 32) | 0x80000000u; // addr[56:32] | type=2
            su32x8 g1;
            g1[0] = 1u << 16;                          // data_size = 2 bytes
            g1[1] = 128u << 16;                        // tensor_dim0 = 128
            g1[2] = 64u << 16;                         // tensor_dim1 = 64
            g1[3] = 128u << 16;                        // tile_dim0 = 128
            g1[4] = 64u;                               // tile_dim1 = 64
            g1[5] = (unsigned)K;                       // tensor_dim0_stride = K
            g1[6] = 0u;
            g1[7] = 0u;
            asm volatile("tensor_load_to_lds %0, %1" :: "s"(g0), "s"(g1) : "memory");
            __builtin_amdgcn_s_wait_tensorcnt(0);
        }
        __syncthreads();

        // next A chunk prefetch (global_prefetch_b8)
        if (kc + 128 < K) {
            __builtin_prefetch(&A[(size_t)mrow0 * lda + kc + 128], 0, 0);
            __builtin_prefetch(&A[(size_t)mrow1 * lda + kc + 128], 0, 0);
        }

#pragma unroll
        for (int k0 = 0; k0 < 128; k0 += 32) {
            const v16bf a0 =
                load_frag(A + (size_t)mrow0 * lda + kc + k0 + kbase);
            const v16bf a1 =
                load_frag(A + (size_t)mrow1 * lda + kc + k0 + kbase);
#pragma unroll
            for (int t = 0; t < 4; ++t) {
                const v16bf b =
                    load_frag(wpan + (size_t)(t * 16 + ncol) * 128 + k0 + kbase);
                acc0[t] = __builtin_amdgcn_wmma_f32_16x16x32_bf16(
                    false, a0, false, b, (short)0, acc0[t], false, false);
                acc1[t] = __builtin_amdgcn_wmma_f32_16x16x32_bf16(
                    false, a1, false, b, (short)0, acc1[t], false, false);
            }
        }
    }

#pragma unroll
    for (int t = 0; t < 4; ++t) {
        const int n = n0 + t * 16 + ncol;
        const float bv = bias[n];
#pragma unroll
        for (int r = 0; r < 8; ++r) {
            float v0 = acc0[t][r] + bv;
            float v1 = acc1[t][r] + bv;
            if (RELU) { v0 = fmaxf(v0, 0.f); v1 = fmaxf(v1, 0.f); }
            Y[(size_t)(m0 + rbase + r) * ldy + n]      = (OUT)v0;
            Y[(size_t)(m0 + 16 + rbase + r) * ldy + n] = (OUT)v1;
        }
    }
}

// ---------------------------------------------------------------------------
// Masked attention. qkv: bf16 [B*S][384] (q|k|v). One wave per (qtile, h, b).
// Scores f32 in LDS; softmax f32; probs re-staged as bf16 for the P@V WMMAs.
// ---------------------------------------------------------------------------
__global__ __launch_bounds__(32)
void attn_kernel(const __bf16* __restrict__ qkv,
                 const int* __restrict__ n_pad,
                 const int* __restrict__ picked,
                 const int* __restrict__ skipped,
                 __bf16* __restrict__ out) {     // bf16 [B*S][DIM]
    __shared__ float  sc [16][SS];   // 32 KB scores
    __shared__ __bf16 scb[16][SS];   // 16 KB probabilities (WMMA A layout src)

    const int lane  = threadIdx.x;
    const int qt    = blockIdx.x;
    const int h     = blockIdx.y;
    const int b     = blockIdx.z;
    const int npd   = n_pad[b];
    const int nc    = picked[b] + skipped[b];
    const int kbase = (lane & 16) >> 1;
    const int rbase = kbase;
    const float scale = 0.17677669529663687f;   // 1/sqrt(32)

    const __bf16* base = qkv + (size_t)b * SS * 384;

    // Q fragment (M=queries, K=head dim, contiguous in memory)
    const int qrow = qt * 16 + (lane & 15);
    const v16bf aq = load_frag(base + (size_t)qrow * 384 + h * HD + kbase);

    // scores = Q @ K^T
    for (int kt = 0; kt < SS / 16; ++kt) {
        const int key = kt * 16 + (lane & 15);
        const v16bf bk =
            load_frag(base + (size_t)key * 384 + DIM + h * HD + kbase);
        v8f c = {};
        c = __builtin_amdgcn_wmma_f32_16x16x32_bf16(
            false, aq, false, bk, (short)0, c, false, false);

        const int j = key;
#pragma unroll
        for (int r = 0; r < 8; ++r) {
            const int i = qt * 16 + rbase + r;
            const bool mid_i = (i >= npd) && (i < SS - nc);
            const bool mid_j = (j >= npd) && (j < SS - nc);
            const bool ch_i  = (i >= SS - nc);
            const bool ch_j  = (j >= SS - nc);
            const bool allowed = ((i < npd) && (i == j)) ||
                                 (mid_i && mid_j) || (ch_i && mid_j) ||
                                 (ch_i && ch_j);
            sc[rbase + r][j] = allowed ? c[r] * scale : NEG_INF;
        }
    }
    __syncthreads();

    // f32 softmax per row; normalized probs stored bf16
    for (int ri = 0; ri < 16; ++ri) {
        float mx = NEG_INF;
        for (int j = lane; j < SS; j += 32) mx = fmaxf(mx, sc[ri][j]);
#pragma unroll
        for (int o = 16; o > 0; o >>= 1) mx = fmaxf(mx, __shfl_xor(mx, o, 32));
        float sum = 0.f;
        for (int j = lane; j < SS; j += 32) {
            const float e = __expf(sc[ri][j] - mx);
            sc[ri][j] = e;
            sum += e;
        }
#pragma unroll
        for (int o = 16; o > 0; o >>= 1) sum += __shfl_xor(sum, o, 32);
        const float inv = 1.f / sum;
        for (int j = lane; j < SS; j += 32)
            scb[ri][j] = (__bf16)(sc[ri][j] * inv);
    }
    __syncthreads();

    // O = P @ V (K over 512 keys in steps of 32; N = hd split 16+16)
    v8f co[2] = {};
    for (int kt = 0; kt < SS / 32; ++kt) {
        const v16bf ap = load_frag(&scb[lane & 15][kt * 32 + kbase]);
#pragma unroll
        for (int t = 0; t < 2; ++t) {
            const int d = t * 16 + (lane & 15);
            v16bf bv;
#pragma unroll
            for (int hh = 0; hh < 16; ++hh) {
                const int key = kt * 32 + frag_k(lane, hh);
                bv[hh] = base[(size_t)key * 384 + 2 * DIM + h * HD + d];
            }
            co[t] = __builtin_amdgcn_wmma_f32_16x16x32_bf16(
                false, ap, false, bv, (short)0, co[t], false, false);
        }
    }
#pragma unroll
    for (int t = 0; t < 2; ++t)
#pragma unroll
        for (int r = 0; r < 8; ++r) {
            const int i = qt * 16 + rbase + r;
            const int d = t * 16 + (lane & 15);
            out[((size_t)b * SS + i) * DIM + h * HD + d] = (__bf16)co[t][r];
        }
}

// ---------------------------------------------------------------------------
// x = layernorm(x + res) * g + b ; also refresh the bf16 shadow of x.
// ---------------------------------------------------------------------------
__global__ __launch_bounds__(32)
void add_ln_kernel(float* __restrict__ x,
                   __bf16* __restrict__ xbf,
                   const float* __restrict__ res,
                   const float* __restrict__ g,
                   const float* __restrict__ bta) {
    const int row  = blockIdx.x;
    const int lane = threadIdx.x;
    float v[4];
    float sum = 0.f;
#pragma unroll
    for (int u = 0; u < 4; ++u) {
        const int d = lane + u * 32;
        v[u] = x[(size_t)row * DIM + d] + res[(size_t)row * DIM + d];
        sum += v[u];
    }
#pragma unroll
    for (int o = 16; o > 0; o >>= 1) sum += __shfl_xor(sum, o, 32);
    const float mean = sum * (1.f / DIM);
    float var = 0.f;
#pragma unroll
    for (int u = 0; u < 4; ++u) { const float t = v[u] - mean; var += t * t; }
#pragma unroll
    for (int o = 16; o > 0; o >>= 1) var += __shfl_xor(var, o, 32);
    const float rstd = rsqrtf(var * (1.f / DIM) + LN_EPS);
#pragma unroll
    for (int u = 0; u < 4; ++u) {
        const int d = lane + u * 32;
        const float y = (v[u] - mean) * rstd * g[d] + bta[d];
        x[(size_t)row * DIM + d]   = y;
        xbf[(size_t)row * DIM + d] = (__bf16)y;
    }
}

// ---------------------------------------------------------------------------
// logits = feat @ proj_w^T + proj_b ; softmax over the 2 logits.
// ---------------------------------------------------------------------------
__global__ __launch_bounds__(128)
void head_kernel(const float* __restrict__ feat,
                 const float* __restrict__ pw,
                 const float* __restrict__ pb,
                 float* __restrict__ out) {
    const int idx = blockIdx.x * blockDim.x + threadIdx.x;
    if (idx >= BSROWS) return;
    const float* x = feat + (size_t)idx * DIM;
    float l0 = pb[0], l1 = pb[1];
    for (int d = 0; d < DIM; ++d) {
        l0 += x[d] * pw[d];
        l1 += x[d] * pw[DIM + d];
    }
    const float m  = fmaxf(l0, l1);
    const float e0 = __expf(l0 - m), e1 = __expf(l1 - m);
    const float inv = 1.f / (e0 + e1);
    out[(size_t)idx * 2 + 0] = e0 * inv;
    out[(size_t)idx * 2 + 1] = e1 * inv;
}

// ---------------------------------------------------------------------------
extern "C" void kernel_launch(void* const* d_in, const int* in_sizes, int n_in,
                              void* d_out, int out_size, void* d_ws, size_t ws_size,
                              hipStream_t stream) {
    const int*   tok     = (const int*)d_in[0];
    const int*   npad    = (const int*)d_in[1];
    const int*   picked  = (const int*)d_in[2];
    const int*   skipped = (const int*)d_in[3];
    const int*   nup     = (const int*)d_in[4];
    const float* emb     = (const float*)d_in[5];
    const float* in_w    = (const float*)d_in[6];
    const float* in_b    = (const float*)d_in[7];
    const float* ow      = (const float*)d_in[8];
    const float* ob      = (const float*)d_in[9];
    const float* ln1g    = (const float*)d_in[10];
    const float* ln1b    = (const float*)d_in[11];
    const float* f1w     = (const float*)d_in[12];
    const float* f1b     = (const float*)d_in[13];
    const float* f2w     = (const float*)d_in[14];
    const float* f2b     = (const float*)d_in[15];
    const float* ln2g    = (const float*)d_in[16];
    const float* ln2b    = (const float*)d_in[17];
    const float* pw      = (const float*)d_in[18];
    const float* pb      = (const float*)d_in[19];

    // ---- workspace carve-up (~82 MiB) ----
    char* p = (char*)d_ws;
    float*  feat   = (float*)p;  p += (size_t)BSROWS * DIM * 4;     // 16 MiB
    float*  tmpf   = (float*)p;  p += (size_t)BSROWS * DIM * 4;     // 16 MiB
    __bf16* featbf = (__bf16*)p; p += (size_t)BSROWS * DIM * 2;     //  8 MiB
    __bf16* bigbf  = (__bf16*)p; p += (size_t)BSROWS * FFDIM * 2;   // 32 MiB (qkv ld384 / ff1 ld512)
    __bf16* attnbf = (__bf16*)p; p += (size_t)BSROWS * DIM * 2;     //  8 MiB
    __bf16* winbf  = (__bf16*)p; p += (size_t)NBLOCKS * 3 * DIM * DIM * 2;
    __bf16* woutbf = (__bf16*)p; p += (size_t)NBLOCKS * DIM * DIM * 2;
    __bf16* wf1bf  = (__bf16*)p; p += (size_t)NBLOCKS * FFDIM * DIM * 2;
    __bf16* wf2bf  = (__bf16*)p; p += (size_t)NBLOCKS * DIM * FFDIM * 2;

    // one-shot weight conversion to bf16
    const int nin = NBLOCKS * 3 * DIM * DIM, nout = NBLOCKS * DIM * DIM,
              nf1 = NBLOCKS * FFDIM * DIM,   nf2 = NBLOCKS * DIM * FFDIM;
    cvt_bf16_kernel<<<(nin + 255) / 256, 256, 0, stream>>>(in_w, winbf, nin);
    cvt_bf16_kernel<<<(nout + 255) / 256, 256, 0, stream>>>(ow, woutbf, nout);
    cvt_bf16_kernel<<<(nf1 + 255) / 256, 256, 0, stream>>>(f1w, wf1bf, nf1);
    cvt_bf16_kernel<<<(nf2 + 255) / 256, 256, 0, stream>>>(f2w, wf2bf, nf2);

    embed_kernel<<<BSROWS, DIM, 0, stream>>>(tok, picked, skipped, nup, emb,
                                             feat, featbf);

    const dim3 gdQKV(BSROWS / 128, (3 * DIM) / 64);
    const dim3 gdOUT(BSROWS / 128, DIM / 64);
    const dim3 gdFF1(BSROWS / 128, FFDIM / 64);

    for (int l = 0; l < NBLOCKS; ++l) {
        // QKV -> bigbf (bf16, ld 384)
        gemm_bf16<false, __bf16><<<gdQKV, 128, 0, stream>>>(
            featbf, DIM, winbf + (size_t)l * 3 * DIM * DIM,
            in_b + (size_t)l * 3 * DIM, bigbf, 3 * DIM, DIM);

        // masked MHA -> attnbf (bf16)
        attn_kernel<<<dim3(SS / 16, NHEADS, BB), 32, 0, stream>>>(
            bigbf, npad, picked, skipped, attnbf);

        // out proj -> tmpf (f32, residual input)
        gemm_bf16<false, float><<<gdOUT, 128, 0, stream>>>(
            attnbf, DIM, woutbf + (size_t)l * DIM * DIM,
            ob + (size_t)l * DIM, tmpf, DIM, DIM);

        // feat = LN(feat + tmpf); refresh featbf
        add_ln_kernel<<<BSROWS, 32, 0, stream>>>(
            feat, featbf, tmpf, ln1g + (size_t)l * DIM, ln1b + (size_t)l * DIM);

        // FF1 (+ReLU) -> bigbf (bf16, ld 512)
        gemm_bf16<true, __bf16><<<gdFF1, 128, 0, stream>>>(
            featbf, DIM, wf1bf + (size_t)l * FFDIM * DIM,
            f1b + (size_t)l * FFDIM, bigbf, FFDIM, DIM);

        // FF2 -> tmpf (f32)
        gemm_bf16<false, float><<<gdOUT, 128, 0, stream>>>(
            bigbf, FFDIM, wf2bf + (size_t)l * DIM * FFDIM,
            f2b + (size_t)l * DIM, tmpf, DIM, FFDIM);

        // feat = LN(feat + tmpf); refresh featbf
        add_ln_kernel<<<BSROWS, 32, 0, stream>>>(
            feat, featbf, tmpf, ln2g + (size_t)l * DIM, ln2b + (size_t)l * DIM);
    }

    head_kernel<<<(BSROWS + 127) / 128, 128, 0, stream>>>(feat, pw, pb,
                                                          (float*)d_out);
}
